// DotProductAttention_10548439679043
// MI455X (gfx1250) — matrix operand order repaired and checked by reading, not compiled, not run on hardware
//
#include <hip/hip_runtime.h>
#include <hip/hip_bf16.h>

typedef __attribute__((ext_vector_type(16))) _Float16 v16h;
typedef __attribute__((ext_vector_type(8)))  _Float16 v8h;
typedef __attribute__((ext_vector_type(4)))  _Float16 v4h;
typedef __attribute__((ext_vector_type(2)))  __fp16   v2fp;
typedef __attribute__((ext_vector_type(8)))  float    v8f;
typedef __attribute__((ext_vector_type(4)))  float    v4f;
typedef __attribute__((ext_vector_type(8)))  unsigned int v8u;

#if __has_builtin(__builtin_amdgcn_exp2f)
#define EXP2(x) __builtin_amdgcn_exp2f(x)
#else
#define EXP2(x) exp2f(x)
#endif

#if __has_builtin(__builtin_amdgcn_permlane16_swap)
#define HAVE_PLSWAP 1
#endif

#define B_    16
#define Q_    2048
#define K_    2048
#define D_    64
#define BK    64            // keys staged per block iteration
#define LDSS  72            // LDS row stride in halves
#define WMMA_F16 __builtin_amdgcn_wmma_f32_16x16x32_f16

__device__ __forceinline__ v16h pack16(v8h lo, v8h hi) {
  v16h r;
#pragma unroll
  for (int i = 0; i < 8; ++i) { r[i] = lo[i]; r[i + 8] = hi[i]; }
  return r;
}

__device__ __forceinline__ v16h loadA(const _Float16* p) {
  return pack16(*(const v8h*)p, *(const v8h*)(p + 16));
}

// lo: value held by lane (n, half0) broadcast to both halves; hi: half1's.
__device__ __forceinline__ void halfsplit(unsigned x, unsigned& lo, unsigned& hi,
                                          int h) {
#ifdef HAVE_PLSWAP
  auto r = __builtin_amdgcn_permlane16_swap(x, x, false, false);
  lo = r[0];
  hi = r[1];
#else
  unsigned o = __shfl_xor(x, 16, 32);
  lo = h ? o : x;
  hi = h ? x : o;
#endif
}

__device__ __forceinline__ float xor16_max(float x) {
#ifdef HAVE_PLSWAP
  auto r = __builtin_amdgcn_permlane16_swap(__builtin_bit_cast(unsigned, x),
                                            __builtin_bit_cast(unsigned, x),
                                            false, false);
  return fmaxf(__builtin_bit_cast(float, r[0]), __builtin_bit_cast(float, r[1]));
#else
  return fmaxf(x, __shfl_xor(x, 16, 32));
#endif
}

__device__ __forceinline__ float xor16_sum(float x) {
#ifdef HAVE_PLSWAP
  auto r = __builtin_amdgcn_permlane16_swap(__builtin_bit_cast(unsigned, x),
                                            __builtin_bit_cast(unsigned, x),
                                            false, false);
  return __builtin_bit_cast(float, r[0]) + __builtin_bit_cast(float, r[1]);
#else
  return x + __shfl_xor(x, 16, 32);
#endif
}

// One 64-key block. S^T = K Q^T ; O^T += V^T P^T.  MASKED: partial tail only.
template <bool MASKED>
__device__ __forceinline__ void process_block(
    int k0, int kend, int vlen, int tid, int n, int h,
    const float* __restrict__ kb, const float* __restrict__ vb,
    _Float16* ldsK, _Float16* ldsVt,
    const v16h (&qfB)[2][2],
    v8f (&accT)[2][4], float (&mrow)[2], float (&lrow)[2])
{
  __syncthreads();   // previous block's LDS reads complete
  // ---- stage K rows (packed b64 stores) ----
#pragma unroll
  for (int i = 0; i < 8; ++i) {
    int idx = tid + 128 * i;          // 1024 float4
    int r   = idx >> 4;               // key 0..63
    int c4  = (idx & 15) << 2;        // d 0..60
    v4f kv = *(const v4f*)(kb + (size_t)(k0 + r) * D_ + c4);
    v4h kh;
#pragma unroll
    for (int jj = 0; jj < 4; ++jj) kh[jj] = (_Float16)kv[jj];
    *(v4h*)(&ldsK[r * LDSS + c4]) = kh;
  }
  // ---- stage V transposed via 4x4 register blocks (packed b64 stores) ----
#pragma unroll
  for (int i = 0; i < 2; ++i) {
    int idx = tid + 128 * i;          // 256 4x4 blocks
    int br  = idx >> 4;               // key block
    int bc  = idx & 15;               // d block
    v4f vr[4];
#pragma unroll
    for (int rr = 0; rr < 4; ++rr)
      vr[rr] = *(const v4f*)(vb + (size_t)(k0 + 4 * br + rr) * D_ + 4 * bc);
#pragma unroll
    for (int cc = 0; cc < 4; ++cc) {
      v4h tv;
#pragma unroll
      for (int rr = 0; rr < 4; ++rr) tv[rr] = (_Float16)vr[rr][cc];
      *(v4h*)(&ldsVt[(4 * bc + cc) * LDSS + 4 * br]) = tv;   // [d][key]
    }
  }
  __syncthreads();

  if (k0 + BK < kend) {
    __builtin_prefetch(kb + (size_t)(k0 + BK) * D_ + tid * 32, 0, 1);
    __builtin_prefetch(vb + (size_t)(k0 + BK) * D_ + tid * 32, 0, 1);
  }

  const _Float16* kbase = &ldsK[n * LDSS + 8 * h];
  const _Float16* vbase = &ldsVt[n * LDSS + 8 * h];

  // ---- S^T: double-buffered A-fragments, each reused by both q-subtiles ----
  v8f sT[2][4];
  v16h aK = loadA(kbase);             // (t=0,f=0)
#pragma unroll
  for (int t = 0; t < 4; ++t)
#pragma unroll
    for (int f = 0; f < 2; ++f) {
      v16h nxt = aK;
      if (t * 2 + f < 7) {
        int t2 = (f == 1) ? t + 1 : t;
        int f2 = f ^ 1;
        nxt = loadA(kbase + (t2 * 16) * LDSS + 32 * f2);  // issue next first
      }
      if (f == 0) {
        sT[0][t] = WMMA_F16(false, aK, false, qfB[0][0], (short)0, (v8f){},
                            false, false);
        sT[1][t] = WMMA_F16(false, aK, false, qfB[1][0], (short)0, (v8f){},
                            false, false);
      } else {
        sT[0][t] = WMMA_F16(false, aK, false, qfB[0][1], (short)0, sT[0][t],
                            false, false);
        sT[1][t] = WMMA_F16(false, aK, false, qfB[1][1], (short)0, sT[1][t],
                            false, false);
      }
      aK = nxt;
    }

  if (MASKED) {   // key index = k0 + 16t + 8h + j (independent of lane n)
#pragma unroll
    for (int t = 0; t < 4; ++t)
#pragma unroll
      for (int j = 0; j < 8; ++j) {
        const bool valid = (k0 + 16 * t + 8 * h + j) < vlen;
        sT[0][t][j] = valid ? sT[0][t][j] : -1e30f;
        sT[1][t][j] = valid ? sT[1][t][j] : -1e30f;
      }
  }

  v16h aV = loadA(vbase);   // first V^T fragment: hidden under the softmax

  // ---- per-lane softmax (one query per lane) + P^T B-fragment build ----
  v16h ptB[2][2];
#pragma unroll
  for (int qs = 0; qs < 2; ++qs) {
    float bm = sT[qs][0][0];
#pragma unroll
    for (int t = 0; t < 4; ++t)
#pragma unroll
      for (int j = 0; j < 8; ++j)
        if (t || j) bm = fmaxf(bm, sT[qs][t][j]);
    bm = xor16_max(bm);                            // join the two key-halves

    float mn    = fmaxf(mrow[qs], bm);
    float alpha = EXP2(mrow[qs] - mn);
    mrow[qs]    = mn;

    unsigned ph[4][4];
    float rs = 0.f;
#pragma unroll
    for (int t = 0; t < 4; ++t)
#pragma unroll
      for (int jj = 0; jj < 4; ++jj) {
        float p0 = EXP2(sT[qs][t][2 * jj] - mn);
        float p1 = EXP2(sT[qs][t][2 * jj + 1] - mn);
        rs += p0 + p1;
        v2fp pk = __builtin_amdgcn_cvt_pkrtz(p0, p1);
        ph[t][jj] = __builtin_bit_cast(unsigned, pk);
      }
    rs = xor16_sum(rs);
    lrow[qs] = lrow[qs] * alpha + rs;
#pragma unroll
    for (int dt = 0; dt < 4; ++dt)
#pragma unroll
      for (int j = 0; j < 8; ++j) accT[qs][dt][j] *= alpha;

    unsigned lo[4][4], hi[4][4];
#pragma unroll
    for (int t = 0; t < 4; ++t)
#pragma unroll
      for (int jj = 0; jj < 4; ++jj)
        halfsplit(ph[t][jj], lo[t][jj], hi[t][jj], h);

    // B frag f: e0..7 <- half-h source of tile 2f ; e8..15 <- half-h source of tile 2f+1
#pragma unroll
    for (int f = 0; f < 2; ++f) {
      v8u r;
#pragma unroll
      for (int jj = 0; jj < 4; ++jj) {
        r[jj]     = h ? hi[2 * f][jj] : lo[2 * f][jj];
        r[jj + 4] = h ? hi[2 * f + 1][jj] : lo[2 * f + 1][jj];
      }
      ptB[qs][f] = __builtin_bit_cast(v16h, r);
    }
  }

  // ---- O^T += V^T P^T : double-buffered A, reused by both subtiles ----
#pragma unroll
  for (int dt = 0; dt < 4; ++dt)
#pragma unroll
    for (int f = 0; f < 2; ++f) {
      v16h nxt = aV;
      if (dt * 2 + f < 7) {
        int d2 = (f == 1) ? dt + 1 : dt;
        int f2 = f ^ 1;
        nxt = loadA(vbase + (d2 * 16) * LDSS + 32 * f2);
      }
      accT[0][dt] = WMMA_F16(false, aV, false, ptB[0][f], (short)0,
                             accT[0][dt], false, false);
      accT[1][dt] = WMMA_F16(false, aV, false, ptB[1][f], (short)0,
                             accT[1][dt], false, false);
      aV = nxt;
    }
}

__global__ __launch_bounds__(128)
void fa_fwd_kernel(const float* __restrict__ qg,
                   const float* __restrict__ kg,
                   const float* __restrict__ vg,
                   const int*   __restrict__ vlens,
                   float*       __restrict__ outg)
{
  __shared__ __align__(32) _Float16 ldsK[BK * LDSS];    // [key][d]
  __shared__ __align__(32) _Float16 ldsVt[D_ * LDSS];   // [d][key]

  const int tid  = threadIdx.x;
  const int lane = tid & 31;
  const int wid  = tid >> 5;
  const int n    = lane & 15;
  const int h    = lane >> 4;

  const int b    = blockIdx.x >> 4;     // 16 q-blocks of 128 rows per batch
  const int qblk = blockIdx.x & 15;
  const int q0   = qblk * 128 + wid * 32;   // 4 waves x 32 q rows

  const int   vlen = vlens[b];
  const float SCL  = 0.125f * 1.4426950408889634f;  // 1/sqrt(64)*log2(e)

  const float* qb = qg + (size_t)b * Q_ * D_;
  const float* kb = kg + (size_t)b * K_ * D_;
  const float* vb = vg + (size_t)b * K_ * D_;
  float*       ob = outg + (size_t)b * Q_ * D_;

  // ---- Q^T B-fragments, pre-scaled (2 subtiles x 2 d-chunks) ----
  v16h qfB[2][2];
#pragma unroll
  for (int qs = 0; qs < 2; ++qs)
#pragma unroll
    for (int f = 0; f < 2; ++f) {
      const float* base = qb + (size_t)(q0 + 16 * qs + n) * D_ + 32 * f + 8 * h;
      v16h bq;
#pragma unroll
      for (int c = 0; c < 4; ++c) {
        v4f x = *(const v4f*)(base + 4 * c + (c >> 1) * 8);
#pragma unroll
        for (int i = 0; i < 4; ++i) bq[4 * c + i] = (_Float16)(x[i] * SCL);
      }
      qfB[qs][f] = bq;
    }

  v8f accT[2][4];
#pragma unroll
  for (int qs = 0; qs < 2; ++qs)
#pragma unroll
    for (int dt = 0; dt < 4; ++dt) accT[qs][dt] = (v8f){};
  float mrow[2] = {-1e30f, -1e30f};
  float lrow[2] = {0.f, 0.f};

  const int kfull = vlen & ~(BK - 1);
  const int kend  = (vlen + BK - 1) & ~(BK - 1);

  for (int k0 = 0; k0 < kfull; k0 += BK)
    process_block<false>(k0, kend, vlen, tid, n, h, kb, vb,
                         ldsK, ldsVt, qfB, accT, mrow, lrow);
  if (kfull < kend)
    process_block<true>(kfull, kend, vlen, tid, n, h, kb, vb,
                        ldsK, ldsVt, qfB, accT, mrow, lrow);

  // ---- epilogue: O^T C layout: d = 16dt + 8h + j (contiguous in j) ----
#pragma unroll
  for (int qs = 0; qs < 2; ++qs) {
    float inv = 1.0f / lrow[qs];
    const int row = q0 + 16 * qs + n;
#pragma unroll
    for (int dt = 0; dt < 4; ++dt) {
      v4f o0, o1;
#pragma unroll
      for (int i = 0; i < 4; ++i) {
        o0[i] = accT[qs][dt][i] * inv;
        o1[i] = accT[qs][dt][i + 4] * inv;
      }
      float* op = ob + (size_t)row * D_ + 16 * dt + 8 * h;
      *(v4f*)(op)     = o0;
      *(v4f*)(op + 4) = o1;
    }
  }
}

extern "C" void kernel_launch(void* const* d_in, const int* in_sizes, int n_in,
                              void* d_out, int out_size, void* d_ws, size_t ws_size,
                              hipStream_t stream) {
  const float* q  = (const float*)d_in[0];
  const float* k  = (const float*)d_in[1];
  const float* v  = (const float*)d_in[2];
  const int*   vl = (const int*)d_in[3];
  float*       o  = (float*)d_out;

  dim3 block(128);                 // 4 waves x 32 q rows = 128 rows/block
  dim3 grid(B_ * (Q_ / 128));      // 256 blocks
  hipLaunchKernelGGL(fa_fwd_kernel, grid, block, 0, stream, q, k, v, vl, o);
}
